// Model_68968584839260
// MI455X (gfx1250) — matrix operand order repaired and checked by reading, not compiled, not run on hardware
//
#include <hip/hip_runtime.h>

// =====================================================================
// pointMLP forward for MI455X (gfx1250, wave32, WMMA).
//
// Roofline: ~1.1 TFLOP of 1x1-conv GEMMs vs ~1 GB of traffic (23.3 TB/s)
// => compute bound => every conv runs on v_wmma_f32_16x16x32_bf16
// (bf16 inputs, f32 accumulate). Geometry / stats / pooling stay f32 VALU.
// Workspace use ~80 MB (features, means, knn idx, bf16 weight copies).
// Per-WG LDS up to 133 KB (CDNA5 allows 320 KB per workgroup).
// =====================================================================

#define WG 256

typedef unsigned int   u32;
typedef unsigned short u16;
typedef __attribute__((ext_vector_type(4)))  u32    u32x4;
typedef __attribute__((ext_vector_type(16))) __bf16 v16bf;
typedef __attribute__((ext_vector_type(8)))  float  v8f;

union FragBF { u32x4 q[2]; v16bf v; };

__device__ __forceinline__ u16 f2bf(float x) {           // RNE f32 -> bf16 bits
  u32 u = __float_as_uint(x);
  u32 r = u + 0x7FFFu + ((u >> 16) & 1u);
  return (u16)(r >> 16);
}
__device__ __forceinline__ float bf2f(u16 h) {
  return __uint_as_float(((u32)h) << 16);
}

struct Conv5 { const u16* W[5]; const float* B[5]; };
struct Conv4 { const u16* W[4]; const float* B[4]; };
struct Conv3 { const u16* W[3]; const float* B[3]; };

// ---------------------------------------------------------------------
// One conv step: dst[32 cols][OUT] = act(W[OUT][Kstr] * src^T + bias)
// src: LDS bf16 [32 cols][Kstr] (row = column/point, col = k).
// A-frag (16x32 bf16, ISA layout): lane m=l%16, khalf=l/16,
//   needs K = {8kh..8kh+7} u {16+8kh..16+8kh+7} -> two b128 loads.
// B-frag (32x16 bf16): lane n=l%16, khalf=l/16, same K order as A:
//   K = {8kh..8kh+7} u {16+8kh..16+8kh+7} -> two b128 loads from LDS.
// D/C (16x16 f32): lane n=l%16, VGPR v -> M = v + 8*khalf.
// K multiple of 32 (zero padded), OUT multiple of 16. EXEC uniform.
// ---------------------------------------------------------------------
__device__ __forceinline__ void gemm_step(
    const u16* __restrict__ W, const float* __restrict__ bias,
    const u16* src, u16* dst, float* dstf,
    int K, int Kstr, int OUT, bool relu)
{
  const int lane = threadIdx.x & 31;
  const int wave = threadIdx.x >> 5;
  const int nwav = blockDim.x >> 5;
  const int lm   = lane & 15;
  const int kh   = lane >> 4;
  const int mtiles = OUT >> 4;

  for (int mt = wave; mt < mtiles; mt += nwav) {
    const u16*  wrow = W + (size_t)(mt * 16 + lm) * Kstr + 8 * kh;
    const float* bp  = bias + mt * 16 + 8 * kh;
    for (int nt = 0; nt < 2; ++nt) {
      const u16* srow = src + (size_t)(nt * 16 + lm) * Kstr + 8 * kh;
      v8f acc;
#pragma unroll
      for (int v = 0; v < 8; ++v) acc[v] = bp[v];
      for (int k0 = 0; k0 < K; k0 += 32) {
        FragBF a, b;
        a.q[0] = *(const u32x4*)(wrow + k0);
        a.q[1] = *(const u32x4*)(wrow + k0 + 16);
        b.q[0] = *(const u32x4*)(srow + k0);
        b.q[1] = *(const u32x4*)(srow + k0 + 16);
        __builtin_prefetch((const void*)(wrow + k0 + 32), 0, 0);
        acc = __builtin_amdgcn_wmma_f32_16x16x32_bf16(
                  false, a.v, false, b.v, (short)0, acc, false, false);
      }
      const int col = nt * 16 + lm;
      const int mr  = mt * 16 + 8 * kh;
      if (dstf) {
        float* dr = dstf + (size_t)col * OUT + mr;
#pragma unroll
        for (int v = 0; v < 8; ++v) { float x = acc[v]; if (relu) x = fmaxf(x, 0.f); dr[v] = x; }
      } else {
        u16* dr = dst + (size_t)col * OUT + mr;
#pragma unroll
        for (int v = 0; v < 8; ++v) { float x = acc[v]; if (relu) x = fmaxf(x, 0.f); dr[v] = f2bf(x); }
      }
    }
  }
}

// ---------------------------------------------------------------------
// Embed: F0[b][n][0..63] = relu(We @ xyz + be), F0[b][n][64..66] = xyz
// ---------------------------------------------------------------------
__global__ void embed_kernel(const float* __restrict__ x, const float* __restrict__ W,
                             const float* __restrict__ bias, float* __restrict__ F0, int N)
{
  const int i = blockIdx.x * blockDim.x + threadIdx.x;  // b*N + n
  const int b = i / N, n = i - b * N;
  const float px = x[(size_t)b * 3 * N + n];
  const float py = x[(size_t)b * 3 * N + N + n];
  const float pz = x[(size_t)b * 3 * N + 2 * N + n];
  float* orow = F0 + (size_t)i * 67;
#pragma unroll 8
  for (int c = 0; c < 64; ++c) {
    float v = W[c * 3] * px + W[c * 3 + 1] * py + W[c * 3 + 2] * pz + bias[c];
    orow[c] = fmaxf(v, 0.f);
  }
  orow[64] = px; orow[65] = py; orow[66] = pz;
}

// ---------------------------------------------------------------------
// kNN (k=32, smallest squared distance): xyz tile staged in LDS,
// one thread per (b,s) keeps a sorted 32-list.
// ---------------------------------------------------------------------
__global__ void knn_kernel(const float* __restrict__ F, int L, int C,
                           const int* __restrict__ fps, int S, int* __restrict__ nnidx)
{
  extern __shared__ char smem[];
  float* sxyz = (float*)smem;          // [L][3]
  const int b = blockIdx.y;
  for (int n = threadIdx.x; n < L; n += blockDim.x) {
    const float* r = F + ((size_t)b * L + n) * (C + 3) + C;
    sxyz[n * 3 + 0] = r[0]; sxyz[n * 3 + 1] = r[1]; sxyz[n * 3 + 2] = r[2];
  }
  __syncthreads();
  const int s = blockIdx.x * blockDim.x + threadIdx.x;
  if (s >= S) return;
  const int a = fps[b * S + s];
  const float cx = sxyz[a * 3], cy = sxyz[a * 3 + 1], cz = sxyz[a * 3 + 2];
  float best[32]; int bidx[32];
#pragma unroll
  for (int j = 0; j < 32; ++j) { best[j] = 3.4e38f; bidx[j] = 0; }
  for (int n = 0; n < L; ++n) {
    const float dx = sxyz[n * 3] - cx, dy = sxyz[n * 3 + 1] - cy, dz = sxyz[n * 3 + 2] - cz;
    const float d = dx * dx + dy * dy + dz * dz;
    if (d < best[31]) {
      int j = 31;
      while (j > 0 && best[j - 1] > d) { best[j] = best[j - 1]; bidx[j] = bidx[j - 1]; --j; }
      best[j] = d; bidx[j] = n;
    }
  }
  int* o = nnidx + ((size_t)b * S + s) * 32;
  for (int j = 0; j < 32; ++j) o[j] = bidx[j];
}

// ---------------------------------------------------------------------
// Group stats: per-(s,c) mean over the 32 neighbors + per-batch sum of
// centered squares (for torch-style unbiased std over the whole batch).
// ---------------------------------------------------------------------
__global__ void stats_kernel(const float* __restrict__ F, int L, int C,
                             const int* __restrict__ nnidx, int S,
                             float* __restrict__ mean, float* __restrict__ ssq)
{
  __shared__ int   sidx[32];
  __shared__ float sred[128];
  const int b = blockIdx.y, s = blockIdx.x;
  const int CH = C + 3;
  if (threadIdx.x < 32) sidx[threadIdx.x] = nnidx[((size_t)b * S + s) * 32 + threadIdx.x];
  __syncthreads();
  float acc = 0.f;
  for (int c = threadIdx.x; c < CH; c += blockDim.x) {
    float vals[32]; float sum = 0.f;
#pragma unroll
    for (int j = 0; j < 32; ++j) {
      vals[j] = F[((size_t)b * L + sidx[j]) * CH + c];
      sum += vals[j];
    }
    const float mu = sum * (1.0f / 32.0f);
    mean[((size_t)b * S + s) * CH + c] = mu;
#pragma unroll
    for (int j = 0; j < 32; ++j) { float d = vals[j] - mu; acc += d * d; }
  }
  sred[threadIdx.x] = acc;
  __syncthreads();
  for (int w = 64; w > 0; w >>= 1) {
    if (threadIdx.x < w) sred[threadIdx.x] += sred[threadIdx.x + w];
    __syncthreads();
  }
  if (threadIdx.x == 0) atomicAdd(&ssq[b], sred[0]);
}

__global__ void finalize_kernel(const float* __restrict__ ssq, float* __restrict__ inv,
                                long long cnt)
{
  const int b = threadIdx.x;
  if (b < 16) inv[b] = 1.0f / (sqrtf(ssq[b] / (float)(cnt - 1)) + 1e-5f);
}

__global__ void zero_kernel(float* p, int n)
{
  const int i = blockIdx.x * blockDim.x + threadIdx.x;
  if (i < n) p[i] = 0.f;
}

// ---------------------------------------------------------------------
// Pre-extraction: one workgroup per (b,s) group. Builds the normalized
// [32 x Dpad] group matrix (bf16) in LDS, then transfer + 4 res convs
// through LDS ping-pong, then max over the 32 neighbors.
// ---------------------------------------------------------------------
__global__ __launch_bounds__(WG) void preextract_kernel(
    const float* __restrict__ Fin, float* __restrict__ Fout,
    const int* __restrict__ fps, const int* __restrict__ nnidx,
    const float* __restrict__ mean, const float* __restrict__ invstd,
    const float* __restrict__ alpha, Conv5 A,
    int C, int L, int S, int OUT, int Dpad)
{
  extern __shared__ char smem[];
  u16* buf0 = (u16*)smem;            // [32][Dpad]
  u16* buf1 = buf0 + 32 * Dpad;      // [32][OUT]
  __shared__ int sidx[32];
  __shared__ int sanchor;

  const int b = blockIdx.y, s = blockIdx.x;
  const int CH = C + 3, D = 2 * C + 3;

  if (threadIdx.x < 32) sidx[threadIdx.x] = nnidx[((size_t)b * S + s) * 32 + threadIdx.x];
  if (threadIdx.x == 0) sanchor = fps[b * S + s];
  __syncthreads();

  const float istd = invstd[b];
  const int p  = threadIdx.x >> 3;   // point 0..31
  const int cb = threadIdx.x & 7;
  const float* mrow = mean + ((size_t)b * S + s) * CH;
  const float* prow = Fin + ((size_t)b * L + sidx[p]) * CH;
  const float* arow = Fin + ((size_t)b * L + sanchor) * CH;
  for (int c = cb; c < Dpad; c += 8) {
    float v;
    if (c < CH)     v = alpha[c] * (prow[c] - mrow[c]) * istd;  // normalized feat+xyz
    else if (c < D) v = arow[c - CH];                           // raw anchor feat
    else            v = 0.f;                                    // K zero-padding
    buf0[p * Dpad + c] = f2bf(v);
  }
  __syncthreads();

  gemm_step(A.W[0], A.B[0], buf0, buf1, nullptr, Dpad, Dpad, OUT, true); __syncthreads();
  gemm_step(A.W[1], A.B[1], buf1, buf0, nullptr, OUT,  OUT,  OUT, true); __syncthreads();
  gemm_step(A.W[2], A.B[2], buf0, buf1, nullptr, OUT,  OUT,  OUT, true); __syncthreads();
  gemm_step(A.W[3], A.B[3], buf1, buf0, nullptr, OUT,  OUT,  OUT, true); __syncthreads();
  gemm_step(A.W[4], A.B[4], buf0, buf1, nullptr, OUT,  OUT,  OUT, true); __syncthreads();

  float* orow = Fout + ((size_t)b * S + s) * (OUT + 3);
  for (int c = threadIdx.x; c < OUT; c += blockDim.x) {
    float m = -3.4e38f;
#pragma unroll 4
    for (int pp = 0; pp < 32; ++pp) m = fmaxf(m, bf2f(buf1[pp * OUT + c]));
    orow[c] = m;
  }
  if (threadIdx.x < 3) orow[OUT + threadIdx.x] = arow[C + threadIdx.x];  // new_xyz
}

// ---------------------------------------------------------------------
// Pos blocks: 32 feature columns per workgroup, 4 convs through LDS.
// ---------------------------------------------------------------------
__global__ __launch_bounds__(WG) void posblock_kernel(
    float* __restrict__ F, int S, int OUT, Conv4 A)
{
  extern __shared__ char smem[];
  u16* buf0 = (u16*)smem;
  u16* buf1 = buf0 + 32 * OUT;
  const int b  = blockIdx.y;
  const int s0 = blockIdx.x * 32;
  const int p  = threadIdx.x >> 3, cb = threadIdx.x & 7;
  float* frow = F + ((size_t)b * S + s0 + p) * (OUT + 3);
  for (int c = cb; c < OUT; c += 8) buf0[p * OUT + c] = f2bf(frow[c]);
  __syncthreads();
  gemm_step(A.W[0], A.B[0], buf0, buf1, nullptr, OUT, OUT, OUT, true); __syncthreads();
  gemm_step(A.W[1], A.B[1], buf1, buf0, nullptr, OUT, OUT, OUT, true); __syncthreads();
  gemm_step(A.W[2], A.B[2], buf0, buf1, nullptr, OUT, OUT, OUT, true); __syncthreads();
  gemm_step(A.W[3], A.B[3], buf1, buf0, nullptr, OUT, OUT, OUT, true); __syncthreads();
  for (int c = cb; c < OUT; c += 8) frow[c] = bf2f(buf0[p * OUT + c]);
}

// ---------------------------------------------------------------------
// Classifier: global max pool (S=128) -> 1024->512->256->40 (batch padded
// to 32 columns, last layer padded to 48 rows, f32 logits out).
// ---------------------------------------------------------------------
__global__ __launch_bounds__(WG) void cls_kernel(
    const float* __restrict__ F4, Conv3 A, float* __restrict__ out)
{
  extern __shared__ char smem[];
  u16* buf0 = (u16*)smem;            // [32][1024]
  u16* buf1 = buf0 + 32 * 1024;      // [32][512]
  const int p = threadIdx.x >> 3, cb = threadIdx.x & 7;
  for (int c = cb; c < 1024; c += 8) {
    float m = 0.f;
    if (p < 16) {
      m = -3.4e38f;
      const float* base = F4 + ((size_t)p * 128) * 1027 + c;
      for (int s = 0; s < 128; ++s) m = fmaxf(m, base[(size_t)s * 1027]);
    }
    buf0[p * 1024 + c] = f2bf(m);
  }
  __syncthreads();
  gemm_step(A.W[0], A.B[0], buf0, buf1, nullptr, 1024, 1024, 512, true); __syncthreads();
  gemm_step(A.W[1], A.B[1], buf1, buf0, nullptr, 512,  512,  256, true); __syncthreads();
  float* flog = (float*)buf1;        // [32][48] f32 logits (6 KB, fits in buf1)
  gemm_step(A.W[2], A.B[2], buf0, nullptr, flog, 256, 256, 48, false); __syncthreads();
  for (int i = threadIdx.x; i < 16 * 40; i += blockDim.x)
    out[i] = flog[(i / 40) * 48 + (i % 40)];
}

// ---------------------------------------------------------------------
// Weight prep: f32 [R][Cin] -> bf16 [Rpad][Cstr] zero padded.
// ---------------------------------------------------------------------
__global__ void convert_w_kernel(const float* __restrict__ src, u16* __restrict__ dst,
                                 int R, int Cin, int Rpad, int Cstr)
{
  const int i = blockIdx.x * blockDim.x + threadIdx.x;
  if (i >= Rpad * Cstr) return;
  const int r = i / Cstr, c = i - r * Cstr;
  const float v = (r < R && c < Cin) ? src[(size_t)r * Cin + c] : 0.f;
  dst[i] = f2bf(v);
}

__global__ void pad_bias_kernel(const float* __restrict__ src, float* __restrict__ dst,
                                int n, int npad)
{
  const int i = threadIdx.x;
  if (i < npad) dst[i] = (i < n) ? src[i] : 0.f;
}

// =====================================================================
extern "C" void kernel_launch(void* const* d_in, const int* in_sizes, int n_in,
                              void* d_out, int out_size, void* d_ws, size_t ws_size,
                              hipStream_t stream)
{
  (void)out_size; (void)ws_size;
  const int Bn = 16, N = 2048;
  const int Cs[5] = {64, 128, 256, 512, 1024};
  const int Ls[5] = {2048, 1024, 512, 256, 128};

  // ---- resolve flattened parameter order from in_sizes -------------
  const float* X; const int* FPS[4];
  const float *embW, *embB;
  const float *alpha[4], *tW[4], *tB[4];
  const float *preW[4][4], *preB[4][4], *posW[4][4], *posB[4][4];
  const float *clsW[3], *clsB[3];

  const bool xFirst = (in_sizes[0] == Bn * 3 * N);
  const int  fb = xFirst ? 1 : 0;
  X = (const float*)(xFirst ? d_in[0] : d_in[n_in - 1]);
  for (int i = 0; i < 4; ++i) FPS[i] = (const int*)d_in[fb + i];
  const int  pb = xFirst ? 5 : 4;
  const bool psorted = (in_sizes[pb] != 192);      // 192 => embed.W first (insertion order)
  if (!psorted) {
    int idx = pb;
    embW = (const float*)d_in[idx++]; embB = (const float*)d_in[idx++];
    for (int i = 0; i < 4; ++i) {
      alpha[i] = (const float*)d_in[idx++];
      tW[i] = (const float*)d_in[idx++]; tB[i] = (const float*)d_in[idx++];
      for (int c = 0; c < 4; ++c) { preW[i][c] = (const float*)d_in[idx++]; preB[i][c] = (const float*)d_in[idx++]; }
      for (int c = 0; c < 4; ++c) { posW[i][c] = (const float*)d_in[idx++]; posB[i][c] = (const float*)d_in[idx++]; }
    }
    for (int c = 0; c < 3; ++c) { clsW[c] = (const float*)d_in[idx++]; clsB[c] = (const float*)d_in[idx++]; }
  } else {
    // jax-pytree sorted keys: cls, embed, stages[alpha,pos,pre,transfer]
    int idx = pb;
    for (int c = 0; c < 3; ++c) { clsW[c] = (const float*)d_in[idx++]; clsB[c] = (const float*)d_in[idx++]; }
    embW = (const float*)d_in[idx++]; embB = (const float*)d_in[idx++];
    for (int i = 0; i < 4; ++i) {
      alpha[i] = (const float*)d_in[idx++];
      for (int c = 0; c < 4; ++c) { posW[i][c] = (const float*)d_in[idx++]; posB[i][c] = (const float*)d_in[idx++]; }
      for (int c = 0; c < 4; ++c) { preW[i][c] = (const float*)d_in[idx++]; preB[i][c] = (const float*)d_in[idx++]; }
      tW[i] = (const float*)d_in[idx++]; tB[i] = (const float*)d_in[idx++];
    }
  }

  // ---- workspace carve-up ------------------------------------------
  char* ws = (char*)d_ws;
  size_t off = 0;
  auto carve = [&](size_t bytes) -> char* {
    char* p = ws + off;
    off = (off + bytes + 255) & ~(size_t)255;
    return p;
  };
  float* F[5];
  for (int i = 0; i < 5; ++i) F[i] = (float*)carve((size_t)Bn * Ls[i] * (Cs[i] + 3) * 4);
  float* meanBuf = (float*)carve((size_t)Bn * 512 * 259 * 4);   // max over stages
  int*   knnBuf  = (int*)carve((size_t)Bn * 1024 * 32 * 4);
  float* ssqBuf  = (float*)carve(64 * 4);
  float* invBuf  = (float*)carve(64 * 4);
  u16* wTc[4]; u16* wPreC[4][4]; u16* wPosC[4][4];
  for (int i = 0; i < 4; ++i) {
    const int OUT = Cs[i + 1], D = 2 * Cs[i] + 3, Dpad = (D + 31) / 32 * 32;
    wTc[i] = (u16*)carve((size_t)OUT * Dpad * 2);
    for (int c = 0; c < 4; ++c) wPreC[i][c] = (u16*)carve((size_t)OUT * OUT * 2);
    for (int c = 0; c < 4; ++c) wPosC[i][c] = (u16*)carve((size_t)OUT * OUT * 2);
  }
  u16* wClsC[3];
  wClsC[0] = (u16*)carve((size_t)512 * 1024 * 2);
  wClsC[1] = (u16*)carve((size_t)256 * 512 * 2);
  wClsC[2] = (u16*)carve((size_t)48 * 256 * 2);
  float* b2pad = (float*)carve(48 * 4);

  // ---- weight conversion to padded bf16 ----------------------------
  auto conv = [&](const float* src, u16* dst, int R, int Cin, int Rpad, int Cstr) {
    const int total = Rpad * Cstr;
    convert_w_kernel<<<(total + 255) / 256, 256, 0, stream>>>(src, dst, R, Cin, Rpad, Cstr);
  };
  for (int i = 0; i < 4; ++i) {
    const int OUT = Cs[i + 1], D = 2 * Cs[i] + 3, Dpad = (D + 31) / 32 * 32;
    conv(tW[i], wTc[i], OUT, D, OUT, Dpad);
    for (int c = 0; c < 4; ++c) conv(preW[i][c], wPreC[i][c], OUT, OUT, OUT, OUT);
    for (int c = 0; c < 4; ++c) conv(posW[i][c], wPosC[i][c], OUT, OUT, OUT, OUT);
  }
  conv(clsW[0], wClsC[0], 512, 1024, 512, 1024);
  conv(clsW[1], wClsC[1], 256, 512, 256, 512);
  conv(clsW[2], wClsC[2], 40, 256, 48, 256);
  pad_bias_kernel<<<1, 64, 0, stream>>>(clsB[2], b2pad, 40, 48);

  // ---- forward ------------------------------------------------------
  embed_kernel<<<(Bn * N) / 256, 256, 0, stream>>>(X, embW, embB, F[0], N);

  for (int i = 0; i < 4; ++i) {
    const int C = Cs[i], L = Ls[i], S = Ls[i + 1], OUT = Cs[i + 1];
    const int D = 2 * C + 3, Dpad = (D + 31) / 32 * 32;

    zero_kernel<<<1, 32, 0, stream>>>(ssqBuf, 16);
    knn_kernel<<<dim3((S + 255) / 256, Bn), 256, (size_t)L * 3 * 4, stream>>>(
        F[i], L, C, FPS[i], S, knnBuf);
    stats_kernel<<<dim3(S, Bn), 128, 0, stream>>>(F[i], L, C, knnBuf, S, meanBuf, ssqBuf);
    finalize_kernel<<<1, 32, 0, stream>>>(ssqBuf, invBuf, (long long)S * 32 * (C + 3));

    Conv5 c5;
    c5.W[0] = wTc[i]; c5.B[0] = tB[i];
    for (int c = 0; c < 4; ++c) { c5.W[1 + c] = wPreC[i][c]; c5.B[1 + c] = preB[i][c]; }
    preextract_kernel<<<dim3(S, Bn), WG, (size_t)32 * (Dpad + OUT) * 2, stream>>>(
        F[i], F[i + 1], FPS[i], knnBuf, meanBuf, invBuf, alpha[i], c5, C, L, S, OUT, Dpad);

    Conv4 c4;
    for (int c = 0; c < 4; ++c) { c4.W[c] = wPosC[i][c]; c4.B[c] = posB[i][c]; }
    posblock_kernel<<<dim3(S / 32, Bn), WG, (size_t)32 * OUT * 4, stream>>>(
        F[i + 1], S, OUT, c4);
  }

  Conv3 c3;
  c3.W[0] = wClsC[0]; c3.B[0] = clsB[0];
  c3.W[1] = wClsC[1]; c3.B[1] = clsB[1];
  c3.W[2] = wClsC[2]; c3.B[2] = b2pad;
  cls_kernel<<<1, WG, (size_t)32 * (1024 + 512) * 2, stream>>>(F[4], c3, (float*)d_out);
}
